// PershomLearnedFilt_6828998001466
// MI455X (gfx1250) — matrix operand order repaired and checked by reading, not compiled, not run on hardware
//
#include <hip/hip_runtime.h>
#include <hip/hip_bf16.h>
#include <math.h>

#define GN_N      262144
#define GN_E      8388608
#define GN_DIM    128
#define GN_K      256          // 2*DIM
#define BN_EPS    1e-5f
#define SLOPE     0.01f

typedef _Float16 h8  __attribute__((ext_vector_type(8)));
typedef _Float16 v16h __attribute__((ext_vector_type(16)));
typedef float    v8f  __attribute__((ext_vector_type(8)));

__device__ __forceinline__ void atomic_add_f32(float* p, float v) {
    (void)__hip_atomic_fetch_add(p, v, __ATOMIC_RELAXED, __HIP_MEMORY_SCOPE_AGENT);
}

__device__ __forceinline__ float leaky(float v) {
    return v > 0.0f ? v : v * SLOPE;
}

// ---------------------------------------------------------------- zero scratch
__global__ void gin_zero_kernel(float* __restrict__ agg,
                                float* __restrict__ s1,
                                float* __restrict__ s2, int nAgg) {
    int i = blockIdx.x * blockDim.x + threadIdx.x;
    if (i < nAgg) agg[i] = 0.0f;
    if (i < 2 * GN_DIM) { s1[i] = 0.0f; s2[i] = 0.0f; }
}

// ------------------------------------------------- fc_W1 -> fp16, transposed
// Wt[c][k] = fc_W1[k][c]  (K-contiguous per output column, for WMMA B frags)
__global__ void gin_wt_kernel(const float* __restrict__ W1,
                              _Float16* __restrict__ Wt) {
    int i = blockIdx.x * blockDim.x + threadIdx.x;   // 0 .. 128*256-1
    if (i >= GN_DIM * GN_K) return;
    int c = i >> 8;          // output column
    int k = i & 255;         // K index
    Wt[i] = (_Float16)W1[(size_t)k * GN_DIM + c];
}

// ---------------------------------------------------------- edge scatter-add
__global__ void gin_edge_kernel(const int* __restrict__ ei,
                                const float* __restrict__ pos,
                                float* __restrict__ agg, int E) {
    int e = blockIdx.x * blockDim.x + threadIdx.x;
    if (e >= E) return;
    int s = ei[e];
    int d = ei[E + e];
    float x = pos[(size_t)s * 3 + 0];
    float y = pos[(size_t)s * 3 + 1];
    float z = pos[(size_t)s * 3 + 2];
    atomic_add_f32(&agg[(size_t)d * 3 + 0], x);
    atomic_add_f32(&agg[(size_t)d * 3 + 1], y);
    atomic_add_f32(&agg[(size_t)d * 3 + 2], z);
}

// ------------------------- y = ((1+eps)*pos + agg) @ W_conv + b ; BN1 stats
// 128 threads/block (thread = feature column), each block walks 256 nodes.
__global__ __launch_bounds__(128)
void gin_conv_kernel(const float* __restrict__ pos,
                     const float* __restrict__ agg,
                     const float* __restrict__ epsp,
                     const float* __restrict__ Wc,   // [3][128]
                     const float* __restrict__ bc,   // [128]
                     _Float16* __restrict__ y16,
                     float* __restrict__ stats1) {
    const int c = threadIdx.x;
    const float w0 = Wc[c], w1 = Wc[GN_DIM + c], w2 = Wc[2 * GN_DIM + c];
    const float b  = bc[c];
    const float sc = 1.0f + epsp[0];
    float s = 0.0f, ss = 0.0f;
    const int n0 = blockIdx.x * 256;
    for (int i = 0; i < 256; ++i) {
        const int n = n0 + i;
        const float x0 = sc * pos[(size_t)n * 3 + 0] + agg[(size_t)n * 3 + 0];
        const float x1 = sc * pos[(size_t)n * 3 + 1] + agg[(size_t)n * 3 + 1];
        const float x2 = sc * pos[(size_t)n * 3 + 2] + agg[(size_t)n * 3 + 2];
        const float yv = fmaf(x0, w0, fmaf(x1, w1, fmaf(x2, w2, b)));
        y16[(size_t)n * GN_DIM + c] = (_Float16)yv;
        s  += yv;
        ss += yv * yv;
    }
    atomic_add_f32(&stats1[c], s);
    atomic_add_f32(&stats1[GN_DIM + c], ss);
}

// ---------------------------------------------- BN finalize: scale/shift
__global__ void gin_bnfin_kernel(const float* __restrict__ stats,
                                 const float* __restrict__ gamma,
                                 const float* __restrict__ beta,
                                 float* __restrict__ p, float invN) {
    const int c = threadIdx.x;
    if (c >= GN_DIM) return;
    const float mean = stats[c] * invN;
    const float var  = stats[GN_DIM + c] * invN - mean * mean;
    const float scl  = gamma[c] * rsqrtf(var + BN_EPS);
    p[c] = scl;
    p[GN_DIM + c] = beta[c] - mean * scl;
}

// ---------------- WMMA GEMM: h = [emb | lrelu(BN1(y))] @ fc_W1 + b1 ; stats2
// 256 threads = 8 waves. Wave w owns output columns [16w, 16w+16).
// Block handles 16 row-tiles of 16 nodes (256 nodes/block).
__global__ __launch_bounds__(256)
void gin_gemm_kernel(const _Float16* __restrict__ y16,
                     const _Float16* __restrict__ Wt,     // [128][256] fp16
                     const int* __restrict__ node_deg,
                     const float* __restrict__ embed,     // [64][128]
                     const float* __restrict__ bn1p,      // scale[128],shift[128]
                     const float* __restrict__ b1,        // [128]
                     _Float16* __restrict__ h16,
                     float* __restrict__ stats2) {
    __shared__ _Float16 At[16][GN_K];                      // 8 KB A tile

    const int tid  = threadIdx.x;
    const int lane = tid & 31;
    const int wave = tid >> 5;           // 0..7
    const int colL = lane & 15;          // column within wave tile / A row
    const int half = lane >> 4;          // K-half select
    const int colG = wave * 16 + colL;   // global output column

    // Preload B fragments (8 K-steps, K-contiguous in Wt) -> 64 VGPRs
    v16h bfrag[8];
#pragma unroll
    for (int kk = 0; kk < 8; ++kk)
        bfrag[kk] = *(const v16h*)(Wt + (size_t)colG * GN_K + kk * 32 + half * 16);

    const float bias = b1[colG];
    float s_acc = 0.0f, ss_acc = 0.0f;

    const int tile0 = blockIdx.x * 16;
    for (int t = 0; t < 16; ++t) {
        const int nodeBase = (tile0 + t) * 16;

        // ---- stage A tile in LDS: 256 threads x 16 halves each
        {
            const int r  = tid >> 4;             // 0..15 row
            const int cB = (tid & 15) * 16;      // 0..240 column base
            const int node = nodeBase + r;
            if (cB < GN_DIM) {                   // embedding half
                const int deg = node_deg[node];
                const float* e = embed + (size_t)deg * GN_DIM + cB;
#pragma unroll
                for (int i = 0; i < 16; ++i)
                    At[r][cB + i] = (_Float16)e[i];
            } else {                             // BN1+LeakyReLU(y) half
                const int c0 = cB - GN_DIM;
                const _Float16* yp = y16 + (size_t)node * GN_DIM + c0;
#pragma unroll
                for (int i = 0; i < 16; ++i) {
                    const int c = c0 + i;
                    float v = (float)yp[i] * bn1p[c] + bn1p[GN_DIM + c];
                    At[r][cB + i] = (_Float16)leaky(v);
                }
            }
        }
        __syncthreads();

        // ---- 8 chained WMMAs, K = 256
        v8f cacc = {};
#pragma unroll
        for (int kk = 0; kk < 8; ++kk) {
            const int kb = kk * 32;
            // A frag, this code's contiguous K order: half0 -> K{0..15}, half1 -> K{16..31} (matches B)
            const h8 lo = *(const h8*)(&At[colL][kb + half * 16]);
            const h8 hi = *(const h8*)(&At[colL][kb + half * 16 + 8]);
            const v16h a = __builtin_shufflevector(lo, hi,
                0, 1, 2, 3, 4, 5, 6, 7, 8, 9, 10, 11, 12, 13, 14, 15);
            cacc = __builtin_amdgcn_wmma_f32_16x16x32_f16(
                false, a, false, bfrag[kk], (short)0, cacc, false, false);
        }

        // ---- epilogue: bias, stats, fp16 store (lane's 8 values = one column)
#pragma unroll
        for (int v = 0; v < 8; ++v) {
            const float hv = cacc[v] + bias;
            s_acc  += hv;
            ss_acc += hv * hv;
            const int row = half * 8 + v;
            h16[(size_t)(nodeBase + row) * GN_DIM + colG] = (_Float16)hv;
        }
        __syncthreads();
    }
    atomic_add_f32(&stats2[colG], s_acc);
    atomic_add_f32(&stats2[GN_DIM + colG], ss_acc);
}

// ------------- final head: BN2 + LeakyReLU + dot(fc_W2) + sigmoid per node
__global__ __launch_bounds__(256)
void gin_final_kernel(const _Float16* __restrict__ h16,
                      const float* __restrict__ bn2p,
                      const float* __restrict__ W2,    // [128]
                      const float* __restrict__ b2p,   // [1]
                      float* __restrict__ out, int N) {
    const int n = blockIdx.x * blockDim.x + threadIdx.x;
    if (n >= N) return;
    const _Float16* hp = h16 + (size_t)n * GN_DIM;
    float acc = 0.0f;
#pragma unroll 4
    for (int c = 0; c < GN_DIM; c += 8) {
        const h8 hv = *(const h8*)(hp + c);
#pragma unroll
        for (int i = 0; i < 8; ++i) {
            float v = (float)hv[i] * bn2p[c + i] + bn2p[GN_DIM + c + i];
            acc += leaky(v) * W2[c + i];
        }
    }
    acc += b2p[0];
    out[n] = 1.0f / (1.0f + __expf(-acc));
}

// ============================================================== launch glue
extern "C" void kernel_launch(void* const* d_in, const int* in_sizes, int n_in,
                              void* d_out, int out_size, void* d_ws, size_t ws_size,
                              hipStream_t stream) {
    const int*   node_deg = (const int*)  d_in[0];
    const int*   edge_idx = (const int*)  d_in[1];
    const float* pos      = (const float*)d_in[2];
    const float* embed    = (const float*)d_in[3];
    const float* epsp     = (const float*)d_in[4];
    const float* W_conv   = (const float*)d_in[5];
    const float* b_conv   = (const float*)d_in[6];
    const float* bn1_g    = (const float*)d_in[7];
    const float* bn1_b    = (const float*)d_in[8];
    const float* fc_W1    = (const float*)d_in[9];
    const float* fc_b1    = (const float*)d_in[10];
    const float* fc_bn_g  = (const float*)d_in[11];
    const float* fc_bn_b  = (const float*)d_in[12];
    const float* fc_W2    = (const float*)d_in[13];
    const float* fc_b2    = (const float*)d_in[14];
    float* out = (float*)d_out;

    // workspace layout (all 256B-aligned)
    char* ws = (char*)d_ws;
    float*    agg    = (float*)   (ws + 0);                    //  3 MB
    _Float16* y16    = (_Float16*)(ws + (size_t)3145728);      // 64 MB
    _Float16* h16    = (_Float16*)(ws + (size_t)70254592);     // 64 MB
    _Float16* Wt     = (_Float16*)(ws + (size_t)137363456);    // 64 KB
    float*    stats1 = (float*)   (ws + (size_t)137429248);
    float*    stats2 = (float*)   (ws + (size_t)137430272);
    float*    bn1p   = (float*)   (ws + (size_t)137431296);
    float*    bn2p   = (float*)   (ws + (size_t)137432320);

    const float invN = 1.0f / (float)GN_N;

    // 1) zero agg + stats
    {
        int nAgg = GN_N * 3;
        int grid = (nAgg + 255) / 256;
        gin_zero_kernel<<<grid, 256, 0, stream>>>(agg, stats1, stats2, nAgg);
    }
    // 2) fc_W1 -> fp16 transposed
    gin_wt_kernel<<<(GN_DIM * GN_K + 255) / 256, 256, 0, stream>>>(fc_W1, Wt);
    // 3) edge scatter-add (L2 fp atomics)
    gin_edge_kernel<<<GN_E / 256, 256, 0, stream>>>(edge_idx, pos, agg, GN_E);
    // 4) GIN conv matvec + BN1 stats
    gin_conv_kernel<<<GN_N / 256, 128, 0, stream>>>(pos, agg, epsp, W_conv,
                                                    b_conv, y16, stats1);
    // 5) BN1 finalize
    gin_bnfin_kernel<<<1, 128, 0, stream>>>(stats1, bn1_g, bn1_b, bn1p, invN);
    // 6) WMMA GEMM (xc @ fc_W1 + b1) + BN2 stats
    gin_gemm_kernel<<<GN_N / 256, 256, 0, stream>>>(y16, Wt, node_deg, embed,
                                                    bn1p, fc_b1, h16, stats2);
    // 7) BN2 finalize
    gin_bnfin_kernel<<<1, 128, 0, stream>>>(stats2, fc_bn_g, fc_bn_b, bn2p, invN);
    // 8) head: BN2 + LeakyReLU + fc_W2 + sigmoid
    gin_final_kernel<<<GN_N / 256, 256, 0, stream>>>(h16, bn2p, fc_W2, fc_b2,
                                                     out, GN_N);
}
